// M_40879498728281
// MI455X (gfx1250) — matrix operand order repaired and checked by reading, not compiled, not run on hardware
//
#include <hip/hip_runtime.h>
#include <hip/hip_bf16.h>

// LayerNorm(512) -> Linear(512->2048, bias) -> exact GELU
// M = 128*14*14 = 25088 rows, K = 512, N = 2048.
// bf16 WMMA GEMM with Tensor-Data-Mover (TDM) LDS staging, double-buffered
// along K, s_wait_tensorcnt sync, nontemporal f32 output stores.

#define D_IN   512
#define D_OUT  2048
#define NROWS  25088   // 128*14*14

// GEMM tiling
#define BLK_M    32          // block M tile (one wave handles all 32 rows)
#define BLK_N    512         // block N tile (8 waves x 64 cols)
#define KC       64          // K chunk staged per TDM transfer
#define NCHUNK   (D_IN / KC) // 8

// LDS layout (bytes). TDM inserts padding: A rows 1024B + 16B pad = 1040;
// B rows 128B + 16B pad = 144.
#define A_ROW_B  1040
#define A_OFF    0
#define A_BYTES  (BLK_M * A_ROW_B)          // 33280
#define B_ROW_B  144
#define B_BYTES  (BLK_N * B_ROW_B)          // 73728
#define B0_OFF   A_BYTES                    // 33280
#define B1_OFF   (A_BYTES + B_BYTES)        // 107008
#define LDS_TOTAL (A_BYTES + 2 * B_BYTES)   // 180736 (<= 320KB WGP LDS)

typedef __attribute__((ext_vector_type(16))) __bf16 v16bf;
typedef __attribute__((ext_vector_type(8)))  float  v8f;
typedef __attribute__((ext_vector_type(4)))  unsigned int u32x4;
typedef __attribute__((ext_vector_type(8)))  int    i32x8;
typedef __attribute__((ext_vector_type(4)))  int    i32x4;

__device__ __forceinline__ unsigned pack_bf16x2(float a, float b) {
    __bf16 x = (__bf16)a;          // RNE conversion
    __bf16 y = (__bf16)b;
    unsigned short ux = __builtin_bit_cast(unsigned short, x);
    unsigned short uy = __builtin_bit_cast(unsigned short, y);
    return (unsigned)ux | ((unsigned)uy << 16);
}

// ---------------------------------------------------------------------------
// TDM: 2D tile load (bf16 elements) global -> LDS with per-row padding.
// D# layout per CDNA5 ISA §8 (group0 128b, group1 256b, groups 2/3 zero for 2D).
//   pad_interval code: interval = 2^(code+1) DWORDs; pad_amount code: (code+1) DWORDs
// Toolchain: 6-arg builtin (u32x4, i32x8, i32x4, i32x4, i32x8, i32 cpol).
// ---------------------------------------------------------------------------
__device__ __forceinline__ void tdm_load_2d_bf16(
    unsigned lds_off, const void* gaddr,
    unsigned tile_d0, unsigned tile_d1,
    unsigned tensor_d0, unsigned tensor_d1,
    unsigned stride0_elems,
    unsigned pad_interval_code, unsigned pad_amount_code)
{
    unsigned long long ga = (unsigned long long)(uintptr_t)gaddr;
    u32x4 g0;
    g0[0] = 1u;                                             // count=1 (valid user D#)
    g0[1] = lds_off;                                        // lds_addr (bytes)
    g0[2] = (unsigned)(ga & 0xFFFFFFFFu);                   // global_addr[31:0]
    g0[3] = (unsigned)((ga >> 32) & 0x1FFFFFFu) | (2u << 30); // addr[56:32] | type=2

    i32x8 g1;
    g1[0] = (int)((1u << 16)                                // data_size = 1 -> 2 bytes
                | (1u << 20)                                // pad_enable
                | (pad_interval_code << 22)
                | (pad_amount_code   << 25));
    g1[1] = (int)((tensor_d0 & 0xFFFFu) << 16);             // tensor_dim0[15:0] @ bit48
    g1[2] = (int)((tensor_d0 >> 16) | ((tensor_d1 & 0xFFFFu) << 16));
    g1[3] = (int)((tensor_d1 >> 16) | (tile_d0 << 16));     // tile_dim0 @ bit112
    g1[4] = (int)(tile_d1);                                 // tile_dim1; tile_dim2=0
    g1[5] = (int)stride0_elems;                             // tensor_dim0_stride[31:0]
    g1[6] = 0;                                              // stride0[47:32], stride1[15:0]
    g1[7] = 0;                                              // stride1[47:16]

    i32x4 g2 = {0, 0, 0, 0};
    i32x4 g3 = {0, 0, 0, 0};
    i32x8 g4 = {0, 0, 0, 0, 0, 0, 0, 0};
    __builtin_amdgcn_tensor_load_to_lds(g0, g1, g2, g3, g4, 0);
}

// ---------------------------------------------------------------------------
// Kernel 1: LayerNorm per row (512 elems), one wave32 per row, output bf16.
// ---------------------------------------------------------------------------
__global__ __launch_bounds__(256) void ln_to_bf16_kernel(
    const float* __restrict__ x,
    const float* __restrict__ gamma,
    const float* __restrict__ beta,
    __bf16* __restrict__ xn)
{
    const int wave = threadIdx.x >> 5;
    const int lane = threadIdx.x & 31;
    const int row  = blockIdx.x * 8 + wave;

    const float* xr = x + (size_t)row * D_IN + lane * 16;
    float4 v[4];
    v[0] = ((const float4*)xr)[0];
    v[1] = ((const float4*)xr)[1];
    v[2] = ((const float4*)xr)[2];
    v[3] = ((const float4*)xr)[3];

    float s = 0.f, ss = 0.f;
#pragma unroll
    for (int i = 0; i < 4; ++i) {
        s  += v[i].x + v[i].y + v[i].z + v[i].w;
        ss += v[i].x * v[i].x + v[i].y * v[i].y + v[i].z * v[i].z + v[i].w * v[i].w;
    }
#pragma unroll
    for (int off = 16; off > 0; off >>= 1) {
        s  += __shfl_xor(s,  off, 32);
        ss += __shfl_xor(ss, off, 32);
    }
    const float mean = s * (1.0f / D_IN);
    const float var  = ss * (1.0f / D_IN) - mean * mean;
    const float inv  = rsqrtf(var + 1e-6f);

    const float* gp = gamma + lane * 16;
    const float* bp = beta  + lane * 16;
    float4 g[4], b[4];
#pragma unroll
    for (int i = 0; i < 4; ++i) { g[i] = ((const float4*)gp)[i]; b[i] = ((const float4*)bp)[i]; }

    uint4 o[2];
    unsigned* op = (unsigned*)o;
#pragma unroll
    for (int i = 0; i < 4; ++i) {
        float r0 = (v[i].x - mean) * inv * g[i].x + b[i].x;
        float r1 = (v[i].y - mean) * inv * g[i].y + b[i].y;
        float r2 = (v[i].z - mean) * inv * g[i].z + b[i].z;
        float r3 = (v[i].w - mean) * inv * g[i].w + b[i].w;
        op[2 * i + 0] = pack_bf16x2(r0, r1);
        op[2 * i + 1] = pack_bf16x2(r2, r3);
    }
    __bf16* dst = xn + (size_t)row * D_IN + lane * 16;
    ((uint4*)dst)[0] = o[0];
    ((uint4*)dst)[1] = o[1];
}

// ---------------------------------------------------------------------------
// Kernel 2: W (f32, [2048,512] row-major) -> bf16.
// ---------------------------------------------------------------------------
__global__ __launch_bounds__(256) void wconv_kernel(
    const float* __restrict__ W, __bf16* __restrict__ Wb)
{
    size_t i = ((size_t)blockIdx.x * 256 + threadIdx.x) * 8;
    float4 a = *(const float4*)(W + i);
    float4 c = *(const float4*)(W + i + 4);
    uint4 o;
    o.x = pack_bf16x2(a.x, a.y);
    o.y = pack_bf16x2(a.z, a.w);
    o.z = pack_bf16x2(c.x, c.y);
    o.w = pack_bf16x2(c.z, c.w);
    *(uint4*)(Wb + i) = o;
}

// ---------------------------------------------------------------------------
// Kernel 3: GEMM + bias + exact GELU, TDM-staged LDS, double-buffered K.
// Block: 8 waves; block tile 32(M) x 512(N); wave tile 32 x 64.
// Per K chunk (KC=64): 2 WMMA k-steps x (2 M-frags x 4 N-frags) = 16 WMMA.
//
// Fragment layouts (ISA 7.12.2, 16-bit, wave32):
//  A 16x32: lane<16 -> row=lane,   K = k0+0..7  & k0+16..23
//           lane>=16 -> row=lane-16, K = k0+8..15 & k0+24..31
//  B 32x16: lane -> col=lane%16; lanes 0-15 hold K=k0+0..7 & k0+16..23,
//           lanes 16-31 hold K=k0+8..15 & k0+24..31.
// ---------------------------------------------------------------------------
__global__ __launch_bounds__(256) void gemm_gelu_kernel(
    const __bf16* __restrict__ A,      // [25088, 512] bf16 (xn)
    const __bf16* __restrict__ B,      // [2048, 512] bf16 (W)
    const float*  __restrict__ bias,   // [2048]
    float* __restrict__ out)           // [25088, 2048]
{
    extern __shared__ char smem[];

    const int lane = threadIdx.x & 31;
    const int wave = threadIdx.x >> 5;
    const int half = lane >> 4;        // 0: lanes 0-15, 1: lanes 16-31
    const int l16  = lane & 15;

    const int mBase  = blockIdx.y * BLK_M;
    const int nBlock = blockIdx.x * BLK_N;
    const int nWave  = nBlock + wave * 64;

    // ---- prologue: TDM A tile (full K) + B chunk 0 ----
    if (wave == 0) {
        // A: 32 rows x 512 elems, row 1024B, pad 16B -> interval code 7 (256 DW),
        // amount code 3 (4 DW)
        tdm_load_2d_bf16(A_OFF, A + (size_t)mBase * D_IN,
                         /*tile*/ D_IN, BLK_M, /*tensor*/ D_IN, BLK_M,
                         /*stride0*/ D_IN, /*pad_i*/ 7, /*pad_a*/ 3);
        // B chunk 0: 512 rows (W cols) x 64 elems, row 128B, pad 16B ->
        // interval code 4 (32 DW), amount code 3 (4 DW)
        tdm_load_2d_bf16(B0_OFF, B + (size_t)nBlock * D_IN,
                         /*tile*/ KC, BLK_N, /*tensor*/ KC, BLK_N,
                         /*stride0*/ D_IN, /*pad_i*/ 4, /*pad_a*/ 3);
    }
    __builtin_amdgcn_s_wait_tensorcnt(0);
    __syncthreads();

    v8f acc[2][4];
#pragma unroll
    for (int t = 0; t < 2; ++t)
#pragma unroll
        for (int nt = 0; nt < 4; ++nt) acc[t][nt] = v8f{};

    const char* aLds = smem + A_OFF;

    for (int c = 0; c < NCHUNK; ++c) {
        // prefetch next B chunk into the other buffer
        if (wave == 0 && (c + 1) < NCHUNK) {
            tdm_load_2d_bf16(((c + 1) & 1) ? B1_OFF : B0_OFF,
                             B + (size_t)nBlock * D_IN + (c + 1) * KC,
                             KC, BLK_N, KC, BLK_N, D_IN, 4, 3);
        }
        const char* bLds = smem + ((c & 1) ? B1_OFF : B0_OFF);

#pragma unroll
        for (int ks = 0; ks < KC / 32; ++ks) {
            const int k0g = c * KC + ks * 32;   // global K for A (full-K LDS tile)
            const int k0b = ks * 32;            // chunk-local K for B

            union { v16bf v; uint4 q[2]; } afr[2];
#pragma unroll
            for (int t = 0; t < 2; ++t) {
                const char* ar = aLds + (size_t)(l16 + t * 16) * A_ROW_B
                               + (size_t)(k0g + half * 8) * 2;
                afr[t].q[0] = *(const uint4*)(ar);
                afr[t].q[1] = *(const uint4*)(ar + 32);
            }
#pragma unroll
            for (int nt = 0; nt < 4; ++nt) {
                union { v16bf v; uint4 q[2]; } bfr;
                const char* br = bLds + (size_t)(wave * 64 + nt * 16 + l16) * B_ROW_B
                               + (size_t)(k0b + half * 8) * 2;
            bfr.q[0] = *(const uint4*)(br);
            bfr.q[1] = *(const uint4*)(br + 32);
                acc[0][nt] = __builtin_amdgcn_wmma_f32_16x16x32_bf16(
                    false, afr[0].v, false, bfr.v, (short)0, acc[0][nt], false, false);
                acc[1][nt] = __builtin_amdgcn_wmma_f32_16x16x32_bf16(
                    false, afr[1].v, false, bfr.v, (short)0, acc[1][nt], false, false);
            }
        }

        __builtin_amdgcn_s_wait_tensorcnt(0);
        __syncthreads();
    }

    // ---- epilogue: bias + exact GELU, nontemporal streaming stores ----
    // C/D layout: lanes 0-15: VGPR j -> (M=j, N=lane); lanes 16-31: (M=8+j, N=lane-16)
#pragma unroll
    for (int t = 0; t < 2; ++t) {
        const int mRow = mBase + t * 16 + half * 8;
#pragma unroll
        for (int nt = 0; nt < 4; ++nt) {
            const int n  = nWave + nt * 16 + l16;
            const float bb = bias[n];
            float* orow = out + (size_t)mRow * D_OUT + n;
#pragma unroll
            for (int j = 0; j < 8; ++j) {
                float h = acc[t][nt][j] + bb;
                float g = 0.5f * h * (1.0f + erff(h * 0.70710678118654752f));
                __builtin_nontemporal_store(g, orow + (size_t)j * D_OUT);
            }
        }
    }
}

// ---------------------------------------------------------------------------
extern "C" void kernel_launch(void* const* d_in, const int* in_sizes, int n_in,
                              void* d_out, int out_size, void* d_ws, size_t ws_size,
                              hipStream_t stream)
{
    const float* x     = (const float*)d_in[0];   // [25088, 512]
    const float* gamma = (const float*)d_in[1];   // [512]
    const float* beta  = (const float*)d_in[2];   // [512]
    const float* W     = (const float*)d_in[3];   // [2048, 512]
    const float* b     = (const float*)d_in[4];   // [2048]
    float* out = (float*)d_out;                   // [25088, 2048]

    // Workspace: xn_bf16 (25.7 MB) | W_bf16 (2 MB)
    __bf16* xn = (__bf16*)d_ws;
    __bf16* Wb = (__bf16*)((char*)d_ws + (size_t)NROWS * D_IN * sizeof(__bf16));

    ln_to_bf16_kernel<<<NROWS / 8, 256, 0, stream>>>(x, gamma, beta, xn);
    wconv_kernel<<<(D_OUT * D_IN) / (256 * 8), 256, 0, stream>>>(W, Wb);

    dim3 grid(D_OUT / BLK_N, NROWS / BLK_M);   // (4, 784)
    gemm_gelu_kernel<<<grid, 256, LDS_TOTAL, stream>>>(xn, Wb, b, out);
}
